// ParallelExperts_29678224015412
// MI455X (gfx1250) — matrix operand order repaired and checked by reading, not compiled, not run on hardware
//
#include <hip/hip_runtime.h>
#include <math.h>

// ---------------- problem constants (from reference) ----------------
#define N_EXPERTS 8
#define TOKENS    8192
#define DMODEL    2048
#define DFF       8192
#define TPE       (TOKENS / N_EXPERTS)   // 1024 tokens per expert

// ---------------- tiling ----------------
#define BM 128
#define BN 128
#define BK 32
#define LDA 40   // halves per sA row (80B stride: pad 32 -> 40 for bank spread)
#define LDB 40   // halves per sB row (N-major, xor-swizzled K within row)

typedef __attribute__((ext_vector_type(16))) _Float16     v16h;
typedef __attribute__((ext_vector_type(4)))  _Float16     v4h;
typedef __attribute__((ext_vector_type(8)))  float        v8f;
typedef __attribute__((ext_vector_type(4)))  float        v4f;
typedef __attribute__((ext_vector_type(4)))  unsigned int v4u;

union FragA { unsigned int u[8]; v16h h; };
union FragB { v4u q[2];          v16h h; };

__device__ __forceinline__ float gelu_tanh(float x) {
  // jax.nn.gelu default (approximate=True)
  float x3 = x * x * x;
  float t  = tanhf(0.7978845608028654f * (x + 0.044715f * x3));
  return 0.5f * x * (1.0f + t);
}

// A fragment (16x32 f16): lane m = lane&15, g = lane>>4, half h -> k = h+8g+(h&8).
// Pairs contiguous -> compiler coalesces into 2 x ds_load_b128.
__device__ __forceinline__ v16h load_fragA(const _Float16* __restrict__ sA,
                                           int mRow, int g) {
  FragA fa;
  const _Float16* p = &sA[mRow * LDA];
#pragma unroll
  for (int j = 0; j < 8; ++j) {
    const int k0 = 2 * j + 8 * g + ((j & 4) << 1);  // {0,2,4,6,16,18,20,22} + 8g
    fa.u[j] = *(const unsigned int*)&p[k0];
  }
  return fa.h;
}

// B fragment (32x16 f16): lane n = lane&15, g = lane>>4, half h -> k = h+8g+(h&8).
// sB is N-major, k xor-swizzled on bits [4:3] -> each chunk is one ds_load_b128.
__device__ __forceinline__ v16h load_fragB(const _Float16* __restrict__ sB,
                                           int n, int g) {
  FragB fb;
  const int sw = ((n >> 4) & 3) << 3;
  const _Float16* p = &sB[n * LDB];
  fb.q[0] = *(const v4u*)&p[(8 * g) ^ sw];
  fb.q[1] = *(const v4u*)&p[(16 + 8 * g) ^ sw];
  return fb.h;
}

// ---------------- kernel 1: H = gelu(X_e @ W1_e), f16 out ----------------
__global__ __launch_bounds__(256)
void moe_up_gelu_k(const float* __restrict__ X,
                   const float* __restrict__ W1,
                   _Float16* __restrict__ H) {
  __shared__ __align__(16) _Float16 sA[BM * LDA];
  __shared__ __align__(16) _Float16 sB[BN * LDB];

  const int e       = blockIdx.z;
  const int rowBase = e * TPE + blockIdx.y * BM;          // token row
  const int colBase = blockIdx.x * BN;                    // dff column
  const float* B = W1 + (size_t)e * DMODEL * DFF;         // [DMODEL][DFF]

  const int tid  = threadIdx.x;
  const int lane = tid & 31;
  const int wave = tid >> 5;
  const int g    = lane >> 4;
  const int wm   = (wave >> 2) * 64;                      // wave M offset
  const int wn   = (wave & 3) * 32;                       // wave N offset

  const int atr = tid >> 3, atc = (tid & 7) * 4;          // A staging coords
  const int btk = tid >> 5, btn = (tid & 31) * 4;         // B staging coords

  const float* pA0 = &X[(size_t)(rowBase + atr) * DMODEL + atc];
  const float* pB0 = &B[(size_t)btk * DFF + colBase + btn];

  // ---- register-pipelined staging: preload tile kt=0 ----
  v4f ra[4], rb[4];
#pragma unroll
  for (int it = 0; it < 4; ++it)
    ra[it] = *(const v4f*)(pA0 + (size_t)(32 * it) * DMODEL);
#pragma unroll
  for (int it = 0; it < 4; ++it)
    rb[it] = *(const v4f*)(pB0 + (size_t)(8 * it) * DFF);

  v8f acc[4][2] = {};

  const int numK = DMODEL / BK;
  for (int kt = 0; kt < numK; ++kt) {
    __syncthreads();   // previous iteration done reading LDS
    // ---- commit staged regs to LDS (f32 -> f16) ----
#pragma unroll
    for (int it = 0; it < 4; ++it) {
      const int r = atr + 32 * it;
      *(v4h*)&sA[r * LDA + atc] =
          (v4h){(_Float16)ra[it].x, (_Float16)ra[it].y,
                (_Float16)ra[it].z, (_Float16)ra[it].w};
    }
#pragma unroll
    for (int it = 0; it < 4; ++it) {
      const int k = btk + 8 * it;
#pragma unroll
      for (int j = 0; j < 4; ++j) {
        const int n  = btn + j;
        const int sw = ((n >> 4) & 3) << 3;
        sB[n * LDB + (k ^ sw)] = (_Float16)rb[it][j];
      }
    }
    __syncthreads();

    // ---- issue next tile's global loads (latency hidden by WMMAs below) ----
    if (kt + 1 < numK) {
      const int kb = (kt + 1) * BK;
#pragma unroll
      for (int it = 0; it < 4; ++it)
        ra[it] = *(const v4f*)(pA0 + kb + (size_t)(32 * it) * DMODEL);
#pragma unroll
      for (int it = 0; it < 4; ++it)
        rb[it] = *(const v4f*)(pB0 + (size_t)(kb + 8 * it) * DFF);
    }
    if (kt + 2 < numK)   // L2 prefetch of the big weight stream
      __builtin_prefetch(pB0 + (size_t)((kt + 2) * BK) * DFF, 0, 1);

    // ---- fragments + 8 WMMAs ----
    v16h fa[4], fb[2];
#pragma unroll
    for (int mt = 0; mt < 4; ++mt)
      fa[mt] = load_fragA(sA, wm + mt * 16 + (lane & 15), g);
#pragma unroll
    for (int nt = 0; nt < 2; ++nt)
      fb[nt] = load_fragB(sB, wn + nt * 16 + (lane & 15), g);
#pragma unroll
    for (int mt = 0; mt < 4; ++mt)
#pragma unroll
      for (int nt = 0; nt < 2; ++nt)
        acc[mt][nt] = __builtin_amdgcn_wmma_f32_16x16x32_f16(
            false, fa[mt], false, fb[nt], (short)0, acc[mt][nt], false, false);
  }

  // ---- epilogue: gelu, f16 store.  C/D layout: vgpr i -> M = i + 8g ----
#pragma unroll
  for (int mt = 0; mt < 4; ++mt)
#pragma unroll
    for (int nt = 0; nt < 2; ++nt)
#pragma unroll
      for (int i = 0; i < 8; ++i) {
        const int m = rowBase + wm + mt * 16 + 8 * g + i;
        const int n = colBase + wn + nt * 16 + (lane & 15);
        H[(size_t)m * DFF + n] = (_Float16)gelu_tanh(acc[mt][nt][i]);
      }
}

// ---------------- kernel 2: Y = H_e @ W2_e, f32 out ----------------
// A tile (f16, no conversion) moved with gfx1250 async global->LDS DMA,
// double-buffered in LDS; B tile register-pipelined with f32->f16 convert.
__global__ __launch_bounds__(256)
void moe_down_k(const _Float16* __restrict__ H,
                const float* __restrict__ W2,
                float* __restrict__ Y) {
  __shared__ __align__(16) _Float16 sA[2][BM * LDA];
  __shared__ __align__(16) _Float16 sB[BN * LDB];

  const int e       = blockIdx.z;
  const int rowBase = e * TPE + blockIdx.y * BM;
  const int colBase = blockIdx.x * BN;
  const float* B = W2 + (size_t)e * DFF * DMODEL;          // [DFF][DMODEL]

  const int tid  = threadIdx.x;
  const int lane = tid & 31;
  const int wave = tid >> 5;
  const int g    = lane >> 4;
  const int wm   = (wave >> 2) * 64;
  const int wn   = (wave & 3) * 32;

  const int atr = tid >> 2, atc8 = (tid & 3) * 8;          // A: 16B f16 chunks
  const int btk = tid >> 5, btn  = (tid & 31) * 4;

  const _Float16* pA0 = &H[(size_t)(rowBase + atr) * DFF + atc8];
  const float*    pB0 = &B[(size_t)btk * DMODEL + colBase + btn];

  // issue async DMA of A tile (2 x b128 per lane) into LDS buffer `buf`
  auto issueA = [&](int kb, int buf) {
#pragma unroll
    for (int it = 0; it < 2; ++it) {
      const int r = atr + 64 * it;
      unsigned ldsa = (unsigned)(uintptr_t)&sA[buf][r * LDA + atc8];  // LDS byte addr = flat[31:0]
      const _Float16* ga = pA0 + kb + (size_t)(64 * it) * DFF;
      asm volatile("global_load_async_to_lds_b128 %0, %1, off"
                   :: "v"(ldsa), "v"(ga) : "memory");
    }
  };

  v4f rb[4];
  auto loadB = [&](int kb) {
#pragma unroll
    for (int it = 0; it < 4; ++it)
      rb[it] = *(const v4f*)(pB0 + (size_t)(kb + 8 * it) * DMODEL);
  };

  issueA(0, 0);
  loadB(0);

  v8f acc[4][2] = {};

  const int numK = DFF / BK;
  for (int kt = 0; kt < numK; ++kt) {
    const int buf = kt & 1;
    __syncthreads();   // all waves done reading sB(kt-1) and sA[buf^1]
    // ---- commit B regs to LDS (f32 -> f16, transposed + swizzled) ----
#pragma unroll
    for (int it = 0; it < 4; ++it) {
      const int k = btk + 8 * it;
#pragma unroll
      for (int j = 0; j < 4; ++j) {
        const int n  = btn + j;
        const int sw = ((n >> 4) & 3) << 3;
        sB[n * LDB + (k ^ sw)] = (_Float16)rb[it][j];
      }
    }
    // ---- issue next tile: async A into other LDS buffer, B into regs ----
    if (kt + 1 < numK) {
      issueA((kt + 1) * BK, buf ^ 1);
      loadB((kt + 1) * BK);
      // 2 async ops of tile kt may still be in flight + 2 just issued:
      // waiting ASYNCcnt<=2 retires tile kt's pair (async loads complete in order)
      asm volatile("s_wait_asynccnt 0x2" ::: "memory");
    } else {
      asm volatile("s_wait_asynccnt 0x0" ::: "memory");
    }
    if (kt + 2 < numK)
      __builtin_prefetch(pB0 + (size_t)((kt + 2) * BK) * DMODEL, 0, 1);
    __syncthreads();   // sA[buf] + sB visible to all waves

    // ---- fragments + 8 WMMAs ----
    v16h fa[4], fb[2];
#pragma unroll
    for (int mt = 0; mt < 4; ++mt)
      fa[mt] = load_fragA(&sA[buf][0], wm + mt * 16 + (lane & 15), g);
#pragma unroll
    for (int nt = 0; nt < 2; ++nt)
      fb[nt] = load_fragB(sB, wn + nt * 16 + (lane & 15), g);
#pragma unroll
    for (int mt = 0; mt < 4; ++mt)
#pragma unroll
      for (int nt = 0; nt < 2; ++nt)
        acc[mt][nt] = __builtin_amdgcn_wmma_f32_16x16x32_f16(
            false, fa[mt], false, fb[nt], (short)0, acc[mt][nt], false, false);
  }

#pragma unroll
  for (int mt = 0; mt < 4; ++mt)
#pragma unroll
    for (int nt = 0; nt < 2; ++nt)
#pragma unroll
      for (int i = 0; i < 8; ++i) {
        const int m = rowBase + wm + mt * 16 + 8 * g + i;
        const int n = colBase + wn + nt * 16 + (lane & 15);
        Y[(size_t)m * DMODEL + n] = acc[mt][nt][i];
      }
}

// ---------------- launch ----------------
extern "C" void kernel_launch(void* const* d_in, const int* in_sizes, int n_in,
                              void* d_out, int out_size, void* d_ws, size_t ws_size,
                              hipStream_t stream) {
  (void)in_sizes; (void)n_in; (void)out_size; (void)ws_size;

  const float* x  = (const float*)d_in[0];
  // d_in[1] = expert_frequency (uniform TOK/E, shapes static) -> unused
  const float* w1 = (const float*)d_in[2];
  const float* w2 = (const float*)d_in[3];
  float*    y     = (float*)d_out;
  _Float16* Hbuf  = (_Float16*)d_ws;   // needs TOKENS*DFF*2 = 128 MiB scratch

  dim3 blk(256, 1, 1);
  dim3 g1(DFF / BN,    TPE / BM, N_EXPERTS);   // (64, 8, 8)
  dim3 g2(DMODEL / BN, TPE / BM, N_EXPERTS);   // (16, 8, 8)

  moe_up_gelu_k<<<g1, blk, 0, stream>>>(x, w1, Hbuf);
  moe_down_k<<<g2, blk, 0, stream>>>(Hbuf, w2, y);
}
